// Attention_37873021616756
// MI455X (gfx1250) — matrix operand order repaired and checked by reading, not compiled, not run on hardware
//
#include <hip/hip_runtime.h>

typedef __bf16 bf16;
typedef __attribute__((ext_vector_type(16))) __bf16 v16bf;
typedef __attribute__((ext_vector_type(8)))  __bf16 v8bf;
typedef __attribute__((ext_vector_type(8)))  float   v8f;

#define WAVES_PER_BLOCK 8
#define THREADS (WAVES_PER_BLOCK * 32)

constexpr int Bn = 4, Sn = 2048, DM = 1024, Hn = 16, DK = 64;
constexpr int S16 = Sn / 16;
constexpr int S32 = Sn / 32;

// Native f32 -> bf16 conversion (gfx1250 has hardware bf16 support).
__device__ __forceinline__ bf16 f2bf(float f) { return (bf16)f; }

// Assemble a 16-half WMMA fragment from two contiguous 16-byte chunks.
__device__ __forceinline__ v16bf ldfrag(const bf16* p0, const bf16* p1) {
  v8bf lo = *(const v8bf*)p0;
  v8bf hi = *(const v8bf*)p1;
  v16bf r;
#pragma unroll
  for (int i = 0; i < 8; ++i) { r[i] = lo[i]; r[i + 8] = hi[i]; }
  return r;
}

__device__ __forceinline__ v8f wmma_bf16(v16bf a, v16bf b, v8f c) {
  return __builtin_amdgcn_wmma_f32_16x16x32_bf16(false, a, false, b, (short)0, c,
                                                 false, false);
}

// ---------------- stage 0: fp32 -> bf16 conversion ----------------
__global__ void cvt_f32_bf16(const float* __restrict__ in, bf16* __restrict__ out, int n) {
  int i = (blockIdx.x * blockDim.x + threadIdx.x) * 4;
  if (i >= n) return;
  float4 v = *(const float4*)(in + i);
  out[i + 0] = f2bf(v.x);
  out[i + 1] = f2bf(v.y);
  out[i + 2] = f2bf(v.z);
  out[i + 3] = f2bf(v.w);
}

// transpose+convert Wq/Wk/Wv [H,DM,DK] f32 -> Wt [3,H,DK,DM] bf16
__global__ void wtrans(const float* __restrict__ Wq, const float* __restrict__ Wk,
                       const float* __restrict__ Wv, bf16* __restrict__ Wt) {
  int idx = blockIdx.x * blockDim.x + threadIdx.x;
  int w = idx / (Hn * DK * DM);
  int rem = idx % (Hn * DK * DM);
  int h = rem / (DK * DM);
  int r2 = rem % (DK * DM);
  int k = r2 / DM;
  int d = r2 % DM;
  const float* src = (w == 0) ? Wq : (w == 1) ? Wk : Wv;
  Wt[idx] = f2bf(src[((size_t)h * DM + d) * DK + k]);
}

// ---------------- stage 1: per-head Q/K/V projections ----------------
// One wave computes a 16x64 tile (16 rows x full d_k) of X[b] @ W[h]:
// the A-fragment is loaded once and reused across 4 B-tiles per 32-deep k step.
// V is written transposed: Vt[b,h, v, s] so PV B-fragments are contiguous.
__global__ void proj_kernel(const bf16* __restrict__ Xq, const bf16* __restrict__ Xk,
                            const bf16* __restrict__ Xv, const bf16* __restrict__ Wt,
                            bf16* __restrict__ Qp, bf16* __restrict__ Kp,
                            bf16* __restrict__ Vt) {
  int t = blockIdx.x * WAVES_PER_BLOCK + (threadIdx.x >> 5);
  int lane = threadIdx.x & 31;
  int m = lane & 15, hw = lane >> 4;
  int st = t % S16; t /= S16;
  int h  = t % Hn;  t /= Hn;
  int b  = t % Bn;  t /= Bn;
  int which = t;  // 0=Q 1=K 2=V
  const bf16* X = (which == 0) ? Xq : (which == 1) ? Xk : Xv;
  const bf16* arow = X + ((size_t)b * Sn + st * 16 + m) * DM;
  const bf16* wbase = Wt + (((size_t)which * Hn + h) * DK) * DM;
  v8f c[4] = {};
#pragma unroll 2
  for (int d0 = 0; d0 < DM; d0 += 32) {
    v16bf a = ldfrag(arow + d0 + hw * 16, arow + d0 + hw * 16 + 8);
#pragma unroll
    for (int nt = 0; nt < 4; ++nt) {
      const bf16* brow = wbase + (size_t)(nt * 16 + m) * DM + d0 + hw * 16;
      c[nt] = wmma_bf16(a, ldfrag(brow, brow + 8), c[nt]);
    }
  }
  size_t bh = (size_t)b * Hn + h;
  if (which == 2) {
#pragma unroll
    for (int nt = 0; nt < 4; ++nt)
#pragma unroll
      for (int r = 0; r < 8; ++r)
        Vt[(bh * DK + nt * 16 + m) * Sn + st * 16 + r + 8 * hw] = f2bf(c[nt][r]);
  } else {
    bf16* dst = (which == 0) ? Qp : Kp;
#pragma unroll
    for (int r = 0; r < 8; ++r) {
      bf16* drow = dst + (bh * Sn + st * 16 + r + 8 * hw) * DK;
#pragma unroll
      for (int nt = 0; nt < 4; ++nt) drow[nt * 16 + m] = f2bf(c[nt][r]);
    }
  }
}

// ---------------- stage 2: flash attention, 32 query rows per wave ----------------
// K/V chunk fragments are loaded once and reused by both 16-row halves;
// softmax row-sums come from an extra WMMA against an all-ones B matrix.
__global__ void attn_kernel(const bf16* __restrict__ Qp, const bf16* __restrict__ Kp,
                            const bf16* __restrict__ Vt, bf16* __restrict__ concat) {
  __shared__ bf16 pbuf[WAVES_PER_BLOCK][32 * 32];  // per-wave prob re-layout buffer
  int wv = threadIdx.x >> 5;
  int lane = threadIdx.x & 31;
  int m = lane & 15, hw = lane >> 4;
  int t = blockIdx.x * WAVES_PER_BLOCK + wv;
  int st = t % S32; t /= S32;
  int h  = t % Hn;  t /= Hn;
  int b  = t;
  size_t bh = (size_t)b * Hn + h;
  const bf16* kbase = Kp + bh * Sn * DK;
  const bf16* vbase = Vt + bh * DK * Sn;

  // Q rows: two row-halves x two k-chunks of 16x32 A-fragments
  v16bf aq[2][2];
#pragma unroll
  for (int si = 0; si < 2; ++si) {
    const bf16* qrow = Qp + (bh * Sn + st * 32 + si * 16 + m) * DK;
    aq[si][0] = ldfrag(qrow + hw * 16,      qrow + hw * 16 + 8);
    aq[si][1] = ldfrag(qrow + 32 + hw * 16, qrow + 40 + hw * 16);
  }
  // all-ones B fragment for row-sum WMMA
  const bf16 one = __builtin_bit_cast(bf16, (unsigned short)0x3F80);
  v16bf ones;
#pragma unroll
  for (int i = 0; i < 16; ++i) ones[i] = one;

  float mrun[2][8], lrun[2][8];
  v8f o[2][4] = {};
#pragma unroll
  for (int si = 0; si < 2; ++si)
#pragma unroll
    for (int r = 0; r < 8; ++r) { mrun[si][r] = -3.0e38f; lrun[si][r] = 0.f; }
  bf16* pb = pbuf[wv];

  for (int t0 = 0; t0 < Sn; t0 += 32) {
    // prefetch next K chunk while this one computes
    __builtin_prefetch(kbase + (size_t)(t0 + 32 + m) * DK + hw * 16, 0, 0);
    // ---- K chunk B-fragments (loaded once, used by both row-halves)
    v16bf bk[2][2];
#pragma unroll
    for (int tt = 0; tt < 2; ++tt) {
      const bf16* kr = kbase + (size_t)(t0 + tt * 16 + m) * DK;
      bk[tt][0] = ldfrag(kr + hw * 16,      kr + hw * 16 + 8);
      bk[tt][1] = ldfrag(kr + 32 + hw * 16, kr + 40 + hw * 16);
    }
    // ---- scores: 4 16x16 tiles, d_k split into 2 WMMAs each
    v8f s[2][2];
#pragma unroll
    for (int si = 0; si < 2; ++si)
#pragma unroll
      for (int tt = 0; tt < 2; ++tt) {
        v8f acc = {};
        acc = wmma_bf16(aq[si][0], bk[tt][0], acc);
        acc = wmma_bf16(aq[si][1], bk[tt][1], acc);
        s[si][tt] = acc;
      }
    // ---- online softmax (row r of half-wave hw lives in VGPR slot r)
#pragma unroll
    for (int si = 0; si < 2; ++si) {
#pragma unroll
      for (int r = 0; r < 8; ++r) {
        float a = s[si][0][r] * 0.125f, bb = s[si][1][r] * 0.125f;  // 1/sqrt(64)
        float cm = fmaxf(a, bb);
        cm = fmaxf(cm, __shfl_xor(cm, 1));
        cm = fmaxf(cm, __shfl_xor(cm, 2));
        cm = fmaxf(cm, __shfl_xor(cm, 4));
        cm = fmaxf(cm, __shfl_xor(cm, 8));
        float mn = fmaxf(mrun[si][r], cm);
        float corr = __expf(mrun[si][r] - mn);
        mrun[si][r] = mn;
        lrun[si][r] *= corr;
#pragma unroll
        for (int nt = 0; nt < 4; ++nt) o[si][nt][r] *= corr;
        // C-layout -> LDS (row-major 32x32 bf16) for A-fragment re-read
        int row = si * 16 + r + 8 * hw;
        pb[row * 32 + m]      = f2bf(__expf(a - mn));
        pb[row * 32 + 16 + m] = f2bf(__expf(bb - mn));
      }
    }
    // ---- P A-fragments from LDS; row sums via WMMA with all-ones B
    v16bf ap[2];
#pragma unroll
    for (int si = 0; si < 2; ++si) {
      const bf16* prow = pb + (size_t)(si * 16 + m) * 32;
      ap[si] = ldfrag(prow + hw * 16, prow + hw * 16 + 8);
      v8f rs = {};
      rs = wmma_bf16(ap[si], ones, rs);  // D[m][*] = rowsum(P), replicated
#pragma unroll
      for (int r = 0; r < 8; ++r) lrun[si][r] += rs[r];
    }
    // ---- P (32x32) @ V-chunk (32x64): V fragments loaded once, used twice
#pragma unroll
    for (int nt = 0; nt < 4; ++nt) {
      const bf16* vrow = vbase + (size_t)(nt * 16 + m) * Sn + t0 + hw * 16;
      v16bf bv = ldfrag(vrow, vrow + 8);
      o[0][nt] = wmma_bf16(ap[0], bv, o[0][nt]);
      o[1][nt] = wmma_bf16(ap[1], bv, o[1][nt]);
    }
  }
  // normalize + store into concat[b, s, h*64 + v] as bf16
#pragma unroll
  for (int si = 0; si < 2; ++si)
#pragma unroll
    for (int r = 0; r < 8; ++r) {
      float inv = 1.0f / lrun[si][r];
      bf16* cr = concat + ((size_t)b * Sn + st * 32 + si * 16 + r + 8 * hw) * DM + h * DK;
#pragma unroll
      for (int nt = 0; nt < 4; ++nt) cr[nt * 16 + m] = f2bf(o[si][nt][r] * inv);
    }
}

// ---------------- stage 3: output projection  out = concat @ Wout^T + bout ----------------
// One wave computes a 16x64 tile; A-fragment reused across 4 B-tiles.
__global__ void outproj_kernel(const bf16* __restrict__ concat, const bf16* __restrict__ WoB,
                               const float* __restrict__ bout, float* __restrict__ out) {
  int t = blockIdx.x * WAVES_PER_BLOCK + (threadIdx.x >> 5);
  int lane = threadIdx.x & 31;
  int m = lane & 15, hw = lane >> 4;
  constexpr int NG = DM / 64;  // 16 groups of 64 output columns
  int ng = t % NG;  t /= NG;
  int st = t % S16; t /= S16;
  int b  = t;
  const bf16* arow = concat + ((size_t)b * Sn + st * 16 + m) * DM;
  const bf16* wbase = WoB + (size_t)(ng * 64) * DM;
  v8f c[4] = {};
#pragma unroll 2
  for (int d0 = 0; d0 < DM; d0 += 32) {
    v16bf a = ldfrag(arow + d0 + hw * 16, arow + d0 + hw * 16 + 8);
#pragma unroll
    for (int nt = 0; nt < 4; ++nt) {
      const bf16* brow = wbase + (size_t)(nt * 16 + m) * DM + d0 + hw * 16;
      c[nt] = wmma_bf16(a, ldfrag(brow, brow + 8), c[nt]);
    }
  }
#pragma unroll
  for (int r = 0; r < 8; ++r) {
    float* orow = out + ((size_t)b * Sn + st * 16 + r + 8 * hw) * DM + ng * 64;
#pragma unroll
    for (int nt = 0; nt < 4; ++nt)
      orow[nt * 16 + m] = c[nt][r] + bout[ng * 64 + nt * 16 + m];
  }
}

extern "C" void kernel_launch(void* const* d_in, const int* in_sizes, int n_in,
                              void* d_out, int out_size, void* d_ws, size_t ws_size,
                              hipStream_t stream) {
  const float* q    = (const float*)d_in[0];
  const float* k    = (const float*)d_in[1];
  const float* v    = (const float*)d_in[2];
  const float* Wq   = (const float*)d_in[3];
  const float* Wk   = (const float*)d_in[4];
  const float* Wv   = (const float*)d_in[5];
  const float* Wout = (const float*)d_in[6];
  const float* bout = (const float*)d_in[7];
  float* out = (float*)d_out;
  (void)in_sizes; (void)n_in; (void)out_size; (void)ws_size;

  char* base = (char*)d_ws;
  size_t off = 0;
  auto carve = [&](size_t elems) {
    bf16* p = (bf16*)(base + off);
    off += elems * sizeof(bf16);
    return p;
  };
  const size_t NX  = (size_t)Bn * Sn * DM;       // 8,388,608
  const size_t NW  = (size_t)3 * Hn * DK * DM;   // 3,145,728
  const size_t NWO = (size_t)DM * DM;            // 1,048,576
  const size_t NP  = (size_t)Bn * Hn * Sn * DK;  // 8,388,608
  bf16* Xq  = carve(NX);
  bf16* Xk  = carve(NX);
  bf16* Xv  = carve(NX);
  bf16* Wt  = carve(NW);
  bf16* WoB = carve(NWO);
  bf16* Qp  = carve(NP);
  bf16* Kp  = carve(NP);
  bf16* Vtr = carve(NP);
  bf16* Cc  = carve(NX);

  cvt_f32_bf16<<<NX / 1024, 256, 0, stream>>>(q, Xq, (int)NX);
  cvt_f32_bf16<<<NX / 1024, 256, 0, stream>>>(k, Xk, (int)NX);
  cvt_f32_bf16<<<NX / 1024, 256, 0, stream>>>(v, Xv, (int)NX);
  cvt_f32_bf16<<<NWO / 1024, 256, 0, stream>>>(Wout, WoB, (int)NWO);
  wtrans<<<NW / 256, 256, 0, stream>>>(Wq, Wk, Wv, Wt);

  int proj_waves = 3 * Bn * Hn * S16;                  // 24576
  proj_kernel<<<proj_waves / WAVES_PER_BLOCK, THREADS, 0, stream>>>(Xq, Xk, Xv, Wt,
                                                                    Qp, Kp, Vtr);
  int attn_waves = Bn * Hn * S32;                      // 4096
  attn_kernel<<<attn_waves / WAVES_PER_BLOCK, THREADS, 0, stream>>>(Qp, Kp, Vtr, Cc);
  int op_waves = Bn * S16 * (DM / 64);                 // 8192
  outproj_kernel<<<op_waves / WAVES_PER_BLOCK, THREADS, 0, stream>>>(Cc, WoB, bout, out);
}
